// Segmenter_33054068310161
// MI455X (gfx1250) — compile-verified, matrix operand order checked by reading, not run on hardware
//
#include <hip/hip_runtime.h>
#include <hip/hip_bf16.h>
#include <math.h>

// ---------------------------------------------------------------------------
// Segmenter (ViT enc/dec) forward on gfx1250.
// All GEMMs run through V_WMMA_F32_16X16X32_BF16 (bf16 in, f32 accum).
// K is a multiple of 32 at every call site (attention scores padded to Sa),
// out-of-range fragment rows are clamped (store-guarded), and the K-loop is
// software-pipelined with double-buffered fragments so global_load_b128 for
// step s+1 overlaps the WMMAs of step s.
// ---------------------------------------------------------------------------

typedef __bf16 bf16_t;
typedef __attribute__((ext_vector_type(16))) __bf16 bf16x16;
typedef __attribute__((ext_vector_type(8)))  float  f32x8;

#define GF_BIAS 1
#define GF_GELU 2
#define GF_RES  4
#define GF_CBT  8   // store bf16 output transposed: Cb[n*ldcb + m]

struct GemmP {
  const bf16_t* A; const bf16_t* W; const float* bias; const float* res;
  float* C; bf16_t* Cb;
  int M, N, K;
  int lda, ldw, ldc, ldcb;
  long sAb, sAh, sWb, sWh, sCb, sCh, sRb, sRh, sBiasH;
  int nh; int flags; float alpha;
};

union FragU { bf16x16 v; uint4 q[2]; };

// One wave computes a 16 x 64 output tile (4 WMMA accumulators).
// A is [M,K] bf16 row-major (lda), W is [N,K] bf16 row-major (ldw) -> C = A*W^T.
// CDNA5 16-bit A-operand striping:
//   lane<16 : K = k0+e (e<8), k0+16+(e-8) (e>=8)
//   lane>=16: K = k0+8+e,     k0+24+(e-8)
__global__ __launch_bounds__(32) void wmma_gemm(GemmP p) {
  const int z  = blockIdx.z;
  const int bi = z / p.nh, hi = z % p.nh;
  const bf16_t* A = p.A + bi * p.sAb + hi * p.sAh;
  const bf16_t* W = p.W + bi * p.sWb + hi * p.sWh;

  const int lane = threadIdx.x;
  const int half = lane >> 4;
  const int l16  = lane & 15;
  const int m0 = blockIdx.x * 16;
  const int n0 = blockIdx.y * 64;

  int am = m0 + l16; if (am >= p.M) am = p.M - 1;   // row clamp (store-guarded)
  const bf16_t* arow = A + (long)am * p.lda + half * 8;

  const bf16_t* wr0; const bf16_t* wr1; const bf16_t* wr2; const bf16_t* wr3;
  {
    int n;
    n = n0 +  0 + l16; if (n >= p.N) n = p.N - 1; wr0 = W + (long)n * p.ldw + half * 8;
    n = n0 + 16 + l16; if (n >= p.N) n = p.N - 1; wr1 = W + (long)n * p.ldw + half * 8;
    n = n0 + 32 + l16; if (n >= p.N) n = p.N - 1; wr2 = W + (long)n * p.ldw + half * 8;
    n = n0 + 48 + l16; if (n >= p.N) n = p.N - 1; wr3 = W + (long)n * p.ldw + half * 8;
  }

  f32x8 acc0 = {0.f,0.f,0.f,0.f,0.f,0.f,0.f,0.f};
  f32x8 acc1 = acc0, acc2 = acc0, acc3 = acc0;

  FragU a0, w00, w01, w02, w03;   // pipeline buffer A
  FragU a1, w10, w11, w12, w13;   // pipeline buffer B

  auto loadA = [&](int kk) {
    a0.q[0] = *reinterpret_cast<const uint4*>(arow + kk);
    a0.q[1] = *reinterpret_cast<const uint4*>(arow + kk + 16);
    w00.q[0] = *reinterpret_cast<const uint4*>(wr0 + kk);
    w00.q[1] = *reinterpret_cast<const uint4*>(wr0 + kk + 16);
    w01.q[0] = *reinterpret_cast<const uint4*>(wr1 + kk);
    w01.q[1] = *reinterpret_cast<const uint4*>(wr1 + kk + 16);
    w02.q[0] = *reinterpret_cast<const uint4*>(wr2 + kk);
    w02.q[1] = *reinterpret_cast<const uint4*>(wr2 + kk + 16);
    w03.q[0] = *reinterpret_cast<const uint4*>(wr3 + kk);
    w03.q[1] = *reinterpret_cast<const uint4*>(wr3 + kk + 16);
  };
  auto loadB = [&](int kk) {
    a1.q[0] = *reinterpret_cast<const uint4*>(arow + kk);
    a1.q[1] = *reinterpret_cast<const uint4*>(arow + kk + 16);
    w10.q[0] = *reinterpret_cast<const uint4*>(wr0 + kk);
    w10.q[1] = *reinterpret_cast<const uint4*>(wr0 + kk + 16);
    w11.q[0] = *reinterpret_cast<const uint4*>(wr1 + kk);
    w11.q[1] = *reinterpret_cast<const uint4*>(wr1 + kk + 16);
    w12.q[0] = *reinterpret_cast<const uint4*>(wr2 + kk);
    w12.q[1] = *reinterpret_cast<const uint4*>(wr2 + kk + 16);
    w13.q[0] = *reinterpret_cast<const uint4*>(wr3 + kk);
    w13.q[1] = *reinterpret_cast<const uint4*>(wr3 + kk + 16);
  };
  auto mmaA = [&]() {
    acc0 = __builtin_amdgcn_wmma_f32_16x16x32_bf16(false, a0.v, false, w00.v, (short)0, acc0, false, false);
    acc1 = __builtin_amdgcn_wmma_f32_16x16x32_bf16(false, a0.v, false, w01.v, (short)0, acc1, false, false);
    acc2 = __builtin_amdgcn_wmma_f32_16x16x32_bf16(false, a0.v, false, w02.v, (short)0, acc2, false, false);
    acc3 = __builtin_amdgcn_wmma_f32_16x16x32_bf16(false, a0.v, false, w03.v, (short)0, acc3, false, false);
  };
  auto mmaB = [&]() {
    acc0 = __builtin_amdgcn_wmma_f32_16x16x32_bf16(false, a1.v, false, w10.v, (short)0, acc0, false, false);
    acc1 = __builtin_amdgcn_wmma_f32_16x16x32_bf16(false, a1.v, false, w11.v, (short)0, acc1, false, false);
    acc2 = __builtin_amdgcn_wmma_f32_16x16x32_bf16(false, a1.v, false, w12.v, (short)0, acc2, false, false);
    acc3 = __builtin_amdgcn_wmma_f32_16x16x32_bf16(false, a1.v, false, w13.v, (short)0, acc3, false, false);
  };

  // software-pipelined over ns = K/32 steps (K always a multiple of 32)
  const int ns = p.K >> 5;
  loadA(0);
  int s = 1;
  for (; s + 1 < ns; s += 2) {
    loadB(s * 32);          // overlap with mmaA
    mmaA();
    loadA((s + 1) * 32);    // overlap with mmaB
    mmaB();
  }
  if (s < ns) {             // even ns: one more pair
    loadB(s * 32);
    mmaA();
    mmaB();
  } else {                  // odd ns: drain
    mmaA();
  }

  const float* bias = (p.flags & GF_BIAS) ? (p.bias + hi * p.sBiasH) : nullptr;
  const float* res  = (p.flags & GF_RES)  ? (p.res + bi * p.sRb + hi * p.sRh) : nullptr;
  float*  C  = p.C  ? (p.C  + bi * p.sCb + hi * p.sCh) : nullptr;
  bf16_t* Cb = p.Cb ? (p.Cb + bi * p.sCb + hi * p.sCh) : nullptr;

  f32x8 accs[4] = {acc0, acc1, acc2, acc3};
  // C/D layout: VGPR r -> M = m0 + r + half*8, N = n0 + t*16 + (lane&15)
  #pragma unroll
  for (int t = 0; t < 4; ++t) {
    const int n = n0 + t * 16 + l16;
    if (n >= p.N) continue;
    const float bv = bias ? bias[n] : 0.f;
    #pragma unroll
    for (int r = 0; r < 8; ++r) {
      const int m = m0 + r + half * 8;
      if (m >= p.M) continue;
      float v = accs[t][r] * p.alpha + bv;
      if (p.flags & GF_GELU) v = 0.5f * v * (1.f + erff(v * 0.70710678118654752f));
      if (res) v += res[(long)m * p.ldc + n];
      if (C)  C[(long)m * p.ldc + n] = v;
      if (Cb) {
        if (p.flags & GF_CBT) Cb[(long)n * p.ldcb + m] = (bf16_t)v;
        else                  Cb[(long)m * p.ldcb + n] = (bf16_t)v;
      }
    }
  }
}

// ------------------------- support kernels ---------------------------------

__global__ void k_transpose(const float* W, bf16_t* Wt, int K, int N,
                            long sW, long sWt) {
  long z = blockIdx.z;
  long i = (long)blockIdx.x * blockDim.x + threadIdx.x;
  if (i >= (long)K * N) return;
  int n = (int)(i / K), k = (int)(i % K);
  Wt[z * sWt + i] = (bf16_t)W[z * sW + (long)k * N + n];
}

__global__ void k_cvt(const float* s, bf16_t* d, long n) {
  long i = (long)blockIdx.x * blockDim.x + threadIdx.x;
  if (i < n) d[i] = (bf16_t)s[i];
}

// images (B,3,224,224) -> patches bf16 (B,196,768); col = c*256 + py*16 + px
__global__ void k_patches(const float* img, bf16_t* out) {
  long i = (long)blockIdx.x * blockDim.x + threadIdx.x;
  if (i >= (long)16 * 196 * 768) return;
  int d = (int)(i % 768); long t = i / 768;
  int pp = (int)(t % 196); int b = (int)(t / 196);
  int c = d / 256, rr = d % 256, py = rr / 16, px = rr % 16;
  int pi = pp / 14, pj = pp % 14;
  out[i] = (bf16_t)img[(((long)b * 3 + c) * 224 + pi * 16 + py) * 224 + pj * 16 + px];
}

// x(B,197,768): s==0 gets class token; add sinusoidal pos-emb everywhere
__global__ void k_posadd(float* x, const float* cls) {
  long i = (long)blockIdx.x * blockDim.x + threadIdx.x;
  if (i >= (long)16 * 197 * 768) return;
  int d = (int)(i % 768); long t = i / 768;
  int s = (int)(t % 197);
  float base = (s == 0) ? cls[d] : x[i];
  float e = ((float)d - (float)(d & 1)) / 768.f;
  float ang = (float)s / powf(10000.f, e);
  float pos = (d & 1) ? cosf(ang) : sinf(ang);
  x[i] = base + pos;
}

__global__ void k_cls(float* xd, const float* cls_emb) {
  long i = (long)blockIdx.x * blockDim.x + threadIdx.x;
  if (i >= (long)16 * 19 * 768) return;
  int d = (int)(i % 768); long t = i / 768;
  int c = (int)(t % 19); int b = (int)(t / 19);
  xd[((long)b * 215 + 196 + c) * 768 + d] = cls_emb[(long)c * 768 + d];
}

__global__ __launch_bounds__(256) void k_ln(const float* x, const float* w,
                                            const float* b, bf16_t* o, int D) {
  long row = blockIdx.x;
  x += row * D; o += row * D;
  __shared__ float ss[256], ss2[256];
  float s = 0.f, s2 = 0.f;
  for (int i = threadIdx.x; i < D; i += 256) { float v = x[i]; s += v; s2 += v * v; }
  ss[threadIdx.x] = s; ss2[threadIdx.x] = s2;
  __syncthreads();
  for (int st = 128; st > 0; st >>= 1) {
    if ((int)threadIdx.x < st) { ss[threadIdx.x] += ss[threadIdx.x + st];
                                 ss2[threadIdx.x] += ss2[threadIdx.x + st]; }
    __syncthreads();
  }
  float mean = ss[0] / D;
  float var  = ss2[0] / D - mean * mean;
  float rstd = rsqrtf(var + 1e-5f);
  for (int i = threadIdx.x; i < D; i += 256)
    o[i] = (bf16_t)((x[i] - mean) * rstd * w[i] + b[i]);
}

// softmax over valid cols [0,S); zero-fill padding cols [S,Sa) so att@V can
// run with K = Sa (multiple of 32) and no K-tail in the GEMM.
__global__ __launch_bounds__(256) void k_softmax(const float* att, bf16_t* out,
                                                 int S, int Sa) {
  long r = blockIdx.x;           // r = bh*S + s
  long bh = r / S, s = r % S;
  const float* row = att + bh * (long)Sa * Sa + s * (long)Sa;
  bf16_t* orow     = out + bh * (long)Sa * Sa + s * (long)Sa;
  __shared__ float red[256];
  float mx = -1e30f;
  for (int i = threadIdx.x; i < S; i += 256) mx = fmaxf(mx, row[i]);
  red[threadIdx.x] = mx; __syncthreads();
  for (int st = 128; st > 0; st >>= 1) {
    if ((int)threadIdx.x < st) red[threadIdx.x] = fmaxf(red[threadIdx.x], red[threadIdx.x + st]);
    __syncthreads();
  }
  mx = red[0]; __syncthreads();
  float sum = 0.f;
  for (int i = threadIdx.x; i < S; i += 256) sum += __expf(row[i] - mx);
  red[threadIdx.x] = sum; __syncthreads();
  for (int st = 128; st > 0; st >>= 1) {
    if ((int)threadIdx.x < st) red[threadIdx.x] += red[threadIdx.x + st];
    __syncthreads();
  }
  float inv = 1.f / red[0];
  for (int i = threadIdx.x; i < S; i += 256)
    orow[i] = (bf16_t)(__expf(row[i] - mx) * inv);
  for (int i = S + threadIdx.x; i < Sa; i += 256)
    orow[i] = (bf16_t)0.f;
}

__global__ __launch_bounds__(256) void k_l2n(const float* x, bf16_t* o, int D) {
  long row = blockIdx.x;
  x += row * D; o += row * D;
  __shared__ float ss[256];
  float s = 0.f;
  for (int i = threadIdx.x; i < D; i += 256) { float v = x[i]; s += v * v; }
  ss[threadIdx.x] = s; __syncthreads();
  for (int st = 128; st > 0; st >>= 1) {
    if ((int)threadIdx.x < st) ss[threadIdx.x] += ss[threadIdx.x + st];
    __syncthreads();
  }
  float rn = rsqrtf(ss[0]);
  for (int i = threadIdx.x; i < D; i += 256) o[i] = (bf16_t)(x[i] * rn);
}

// LN over 19 classes + transpose (b,14,14,19)->(b,19,14,14)
__global__ void k_masknorm(const float* masks, const float* mw, const float* mb,
                           float* mT) {
  long r = (long)blockIdx.x * blockDim.x + threadIdx.x;
  if (r >= (long)16 * 196) return;
  const float* row = masks + r * 19;
  float s = 0.f, s2 = 0.f;
  for (int c = 0; c < 19; ++c) { float v = row[c]; s += v; s2 += v * v; }
  float mean = s / 19.f, var = s2 / 19.f - mean * mean;
  float rstd = rsqrtf(var + 1e-5f);
  int b = (int)(r / 196), p = (int)(r % 196), pi = p / 14, pj = p % 14;
  for (int c = 0; c < 19; ++c)
    mT[(((long)b * 19 + c) * 14 + pi) * 14 + pj] = (row[c] - mean) * rstd * mw[c] + mb[c];
}

// bilinear 14x14 -> 224x224, half-pixel centers, edge clamp
__global__ void k_resize(const float* mT, float* out) {
  long i = (long)blockIdx.x * blockDim.x + threadIdx.x;
  if (i >= (long)16 * 19 * 224 * 224) return;
  int x = (int)(i % 224); long t = i / 224;
  int y = (int)(t % 224); t /= 224;
  int c = (int)(t % 19);  int b = (int)(t / 19);
  float fy = (y + 0.5f) * (14.f / 224.f) - 0.5f;
  float fx = (x + 0.5f) * (14.f / 224.f) - 0.5f;
  int y0 = (int)floorf(fy); float wy = fy - y0;
  int x0 = (int)floorf(fx); float wx = fx - x0;
  int y0c = y0 < 0 ? 0 : (y0 > 13 ? 13 : y0);
  int y1 = y0 + 1; int y1c = y1 < 0 ? 0 : (y1 > 13 ? 13 : y1);
  int x0c = x0 < 0 ? 0 : (x0 > 13 ? 13 : x0);
  int x1 = x0 + 1; int x1c = x1 < 0 ? 0 : (x1 > 13 ? 13 : x1);
  const float* base = mT + ((long)b * 19 + c) * 196;
  float v00 = base[y0c * 14 + x0c], v01 = base[y0c * 14 + x1c];
  float v10 = base[y1c * 14 + x0c], v11 = base[y1c * 14 + x1c];
  out[i] = v00 * (1.f - wy) * (1.f - wx) + v01 * (1.f - wy) * wx +
           v10 * wy * (1.f - wx) + v11 * wy * wx;
}

// ------------------------- host orchestration ------------------------------

static const int Bn = 16, Dm = 768, NHn = 12, DHn = 64, NC = 19, MLPn = 3072;
static const int S1 = 197, S2 = 215, SaP = 224, NP = 196;  // SaP: padded S (mult of 32)

static GemmP gdef() { GemmP g{}; g.alpha = 1.f; g.nh = 1; return g; }

static void launch_gemm(const GemmP& p, int batches, hipStream_t st) {
  dim3 g((p.M + 15) / 16, (p.N + 63) / 64, batches);
  hipLaunchKernelGGL(wmma_gemm, g, dim3(32), 0, st, p);
}

struct BlkW {
  const float *ln1w, *ln1b, *qw, *qb, *kw, *kb, *vw, *vb;
  const float *ln2w, *ln2b, *f1w, *f1b, *f2w, *f2b;
};

static void run_block(float* x, int S, const BlkW& Wb,
                      bf16_t* xln, bf16_t* qbuf, bf16_t* kbuf, bf16_t* vT,
                      float* attf, bf16_t* attb, bf16_t* hid,
                      bf16_t* wqT, bf16_t* wkT, bf16_t* wvT,
                      bf16_t* f1T, bf16_t* f2T, hipStream_t st) {
  const int BS = Bn * S;
  const int Sa = SaP;
  // per-layer weight transposes (fp32 -> bf16 [N,K])
  {
    dim3 gq(((long)DHn * DHn + 255) / 256, 1, NHn);
    hipLaunchKernelGGL(k_transpose, gq, dim3(256), 0, st, Wb.qw, wqT, DHn, DHn,
                       (long)DHn * DHn, (long)DHn * DHn);
    hipLaunchKernelGGL(k_transpose, gq, dim3(256), 0, st, Wb.kw, wkT, DHn, DHn,
                       (long)DHn * DHn, (long)DHn * DHn);
    hipLaunchKernelGGL(k_transpose, gq, dim3(256), 0, st, Wb.vw, wvT, DHn, DHn,
                       (long)DHn * DHn, (long)DHn * DHn);
    dim3 g1(((long)Dm * MLPn + 255) / 256, 1, 1);
    hipLaunchKernelGGL(k_transpose, g1, dim3(256), 0, st, Wb.f1w, f1T, Dm, MLPn, 0L, 0L);
    hipLaunchKernelGGL(k_transpose, g1, dim3(256), 0, st, Wb.f2w, f2T, MLPn, Dm, 0L, 0L);
  }
  // ln1
  hipLaunchKernelGGL(k_ln, dim3(BS), dim3(256), 0, st, x, Wb.ln1w, Wb.ln1b, xln, Dm);

  // Q,K,V projections: batched over (b,h)
  {
    GemmP g = gdef();
    g.A = xln; g.lda = Dm; g.sAb = (long)S * Dm; g.sAh = DHn;
    g.ldw = DHn; g.sWh = (long)DHn * DHn;
    g.sBiasH = DHn;
    g.M = S; g.N = DHn; g.K = DHn; g.nh = NHn; g.flags = GF_BIAS;
    // Q
    g.W = wqT; g.bias = Wb.qb; g.Cb = qbuf; g.ldcb = DHn;
    g.sCb = (long)NHn * S * DHn; g.sCh = (long)S * DHn;
    launch_gemm(g, Bn * NHn, st);
    // K
    g.W = wkT; g.bias = Wb.kb; g.Cb = kbuf;
    launch_gemm(g, Bn * NHn, st);
    // V (store transposed [DH, Sa] for att@V)
    g.W = wvT; g.bias = Wb.vb; g.Cb = vT; g.flags = GF_BIAS | GF_CBT;
    g.ldcb = Sa; g.sCb = (long)NHn * DHn * Sa; g.sCh = (long)DHn * Sa;
    launch_gemm(g, Bn * NHn, st);
  }
  // att = Q K^T / 8
  {
    GemmP g = gdef();
    g.A = qbuf; g.lda = DHn; g.sAb = (long)NHn * S * DHn; g.sAh = (long)S * DHn;
    g.W = kbuf; g.ldw = DHn; g.sWb = g.sAb; g.sWh = g.sAh;
    g.C = attf; g.ldc = Sa; g.sCb = (long)NHn * Sa * Sa; g.sCh = (long)Sa * Sa;
    g.M = S; g.N = S; g.K = DHn; g.nh = NHn; g.alpha = 0.125f;
    launch_gemm(g, Bn * NHn, st);
  }
  hipLaunchKernelGGL(k_softmax, dim3(Bn * NHn * S), dim3(256), 0, st, attf, attb, S, Sa);
  // x += att @ V  (K runs over padded Sa; padding cols of att are zero)
  {
    GemmP g = gdef();
    g.A = attb; g.lda = Sa; g.sAb = (long)NHn * Sa * Sa; g.sAh = (long)Sa * Sa;
    g.W = vT; g.ldw = Sa; g.sWb = (long)NHn * DHn * Sa; g.sWh = (long)DHn * Sa;
    g.C = x; g.ldc = Dm; g.sCb = (long)S * Dm; g.sCh = DHn;
    g.res = x; g.sRb = (long)S * Dm; g.sRh = DHn;
    g.M = S; g.N = DHn; g.K = Sa; g.nh = NHn; g.flags = GF_RES;
    launch_gemm(g, Bn * NHn, st);
  }
  // ln2
  hipLaunchKernelGGL(k_ln, dim3(BS), dim3(256), 0, st, x, Wb.ln2w, Wb.ln2b, xln, Dm);
  // fc1 + GELU
  {
    GemmP g = gdef();
    g.A = xln; g.lda = Dm; g.W = f1T; g.ldw = Dm;
    g.bias = Wb.f1b; g.Cb = hid; g.ldcb = MLPn;
    g.M = BS; g.N = MLPn; g.K = Dm; g.flags = GF_BIAS | GF_GELU;
    launch_gemm(g, 1, st);
  }
  // x += fc2(hidden)
  {
    GemmP g = gdef();
    g.A = hid; g.lda = MLPn; g.W = f2T; g.ldw = MLPn;
    g.bias = Wb.f2b; g.C = x; g.ldc = Dm; g.res = x;
    g.M = BS; g.N = Dm; g.K = MLPn; g.flags = GF_BIAS | GF_RES;
    launch_gemm(g, 1, st);
  }
}

static BlkW mk_blk(void* const* d_in, int base, int l) {
  BlkW w;
  const long sD = Dm, sH = (long)NHn * DHn;
  w.ln1w = (const float*)d_in[base + 0] + (long)l * sD;
  w.ln1b = (const float*)d_in[base + 1] + (long)l * sD;
  w.qw = (const float*)d_in[base + 2] + (long)l * NHn * DHn * DHn;
  w.qb = (const float*)d_in[base + 3] + (long)l * sH;
  w.kw = (const float*)d_in[base + 4] + (long)l * NHn * DHn * DHn;
  w.kb = (const float*)d_in[base + 5] + (long)l * sH;
  w.vw = (const float*)d_in[base + 6] + (long)l * NHn * DHn * DHn;
  w.vb = (const float*)d_in[base + 7] + (long)l * sH;
  w.ln2w = (const float*)d_in[base + 8] + (long)l * sD;
  w.ln2b = (const float*)d_in[base + 9] + (long)l * sD;
  w.f1w = (const float*)d_in[base + 10] + (long)l * Dm * MLPn;
  w.f1b = (const float*)d_in[base + 11] + (long)l * MLPn;
  w.f2w = (const float*)d_in[base + 12] + (long)l * MLPn * Dm;
  w.f2b = (const float*)d_in[base + 13] + (long)l * sD;
  return w;
}

extern "C" void kernel_launch(void* const* d_in, const int* in_sizes, int n_in,
                              void* d_out, int out_size, void* d_ws, size_t ws_size,
                              hipStream_t stream) {
  (void)in_sizes; (void)n_in; (void)out_size; (void)ws_size;
  const float* images  = (const float*)d_in[0];
  const float* lm_w    = (const float*)d_in[1];
  const float* lm_b    = (const float*)d_in[2];
  const float* cls_tok = (const float*)d_in[3];
  const float* cls_emb = (const float*)d_in[18];
  const float* pdec_w  = (const float*)d_in[19];
  const float* pdec_b  = (const float*)d_in[20];
  const float* pp_w    = (const float*)d_in[35];
  const float* pc_w    = (const float*)d_in[36];
  const float* dnw     = (const float*)d_in[37];
  const float* dnb     = (const float*)d_in[38];
  const float* mnw     = (const float*)d_in[39];
  const float* mnb     = (const float*)d_in[40];
  float* out = (float*)d_out;

  // workspace bump allocator
  size_t off = 0;
  char* wsb = (char*)d_ws;
  auto alloc = [&](size_t bytes) -> void* {
    void* p = wsb + off;
    off += (bytes + 255) & ~(size_t)255;
    return p;
  };
  const long BS1 = (long)Bn * S1, BS2 = (long)Bn * S2;
  float*  x    = (float*) alloc(BS1 * Dm * 4);
  float*  xd   = (float*) alloc(BS2 * Dm * 4);
  bf16_t* xln  = (bf16_t*)alloc(BS2 * Dm * 2);
  bf16_t* qbuf = (bf16_t*)alloc((long)Bn * NHn * S2 * DHn * 2);
  bf16_t* kbuf = (bf16_t*)alloc((long)Bn * NHn * S2 * DHn * 2);
  bf16_t* vT   = (bf16_t*)alloc((long)Bn * NHn * DHn * SaP * 2);
  float*  attf = (float*) alloc((long)Bn * NHn * SaP * SaP * 4);
  bf16_t* attb = (bf16_t*)alloc((long)Bn * NHn * SaP * SaP * 2);
  bf16_t* hid  = (bf16_t*)alloc(BS2 * MLPn * 2);
  bf16_t* pat  = (bf16_t*)alloc((long)Bn * NP * Dm * 2);
  bf16_t* wqT  = (bf16_t*)alloc((long)NHn * DHn * DHn * 2);
  bf16_t* wkT  = (bf16_t*)alloc((long)NHn * DHn * DHn * 2);
  bf16_t* wvT  = (bf16_t*)alloc((long)NHn * DHn * DHn * 2);
  bf16_t* f1T  = (bf16_t*)alloc((long)Dm * MLPn * 2);
  bf16_t* f2T  = (bf16_t*)alloc((long)MLPn * Dm * 2);
  bf16_t* lmT  = (bf16_t*)alloc((long)Dm * Dm * 2);
  bf16_t* pdT  = (bf16_t*)alloc((long)Dm * Dm * 2);
  bf16_t* ppT  = (bf16_t*)alloc((long)Dm * Dm * 2);
  bf16_t* pcT  = (bf16_t*)alloc((long)Dm * Dm * 2);
  float*  ptf  = (float*) alloc((long)Bn * NP * Dm * 4);
  float*  cff  = (float*) alloc((long)Bn * NC * Dm * 4);
  bf16_t* ptn  = (bf16_t*)alloc((long)Bn * NP * Dm * 2);
  bf16_t* cfn  = (bf16_t*)alloc((long)Bn * NC * Dm * 2);
  float*  mskf = (float*) alloc((long)Bn * NP * NC * 4);
  float*  mskT = (float*) alloc((long)Bn * NC * NP * 4);

  // persistent weight transposes
  {
    dim3 g(((long)Dm * Dm + 255) / 256, 1, 1);
    hipLaunchKernelGGL(k_transpose, g, dim3(256), 0, stream, lm_w,  lmT, Dm, Dm, 0L, 0L);
    hipLaunchKernelGGL(k_transpose, g, dim3(256), 0, stream, pdec_w, pdT, Dm, Dm, 0L, 0L);
    hipLaunchKernelGGL(k_transpose, g, dim3(256), 0, stream, pp_w,  ppT, Dm, Dm, 0L, 0L);
    hipLaunchKernelGGL(k_transpose, g, dim3(256), 0, stream, pc_w,  pcT, Dm, Dm, 0L, 0L);
  }
  // patch extraction + embed (writes x rows s=1..196 per batch)
  hipLaunchKernelGGL(k_patches, dim3(((long)Bn * NP * Dm + 255) / 256), dim3(256),
                     0, stream, images, pat);
  {
    GemmP g = gdef();
    g.A = pat; g.lda = Dm; g.sAb = (long)NP * Dm;
    g.W = lmT; g.ldw = Dm;
    g.bias = lm_b; g.C = x + Dm; g.ldc = Dm; g.sCb = (long)S1 * Dm;
    g.M = NP; g.N = Dm; g.K = Dm; g.flags = GF_BIAS;
    launch_gemm(g, Bn, stream);
  }
  hipLaunchKernelGGL(k_posadd, dim3((BS1 * Dm + 255) / 256), dim3(256), 0, stream,
                     x, cls_tok);

  // encoder
  for (int l = 0; l < 6; ++l) {
    BlkW w = mk_blk(d_in, 4, l);
    run_block(x, S1, w, xln, qbuf, kbuf, vT, attf, attb, hid,
              wqT, wkT, wvT, f1T, f2T, stream);
  }
  // proj to decoder space: drop last token (keep s=0..195)
  hipLaunchKernelGGL(k_cvt, dim3((BS1 * Dm + 255) / 256), dim3(256), 0, stream,
                     x, xln, BS1 * Dm);
  {
    GemmP g = gdef();
    g.A = xln; g.lda = Dm; g.sAb = (long)S1 * Dm;
    g.W = pdT; g.ldw = Dm; g.bias = pdec_b;
    g.C = xd; g.ldc = Dm; g.sCb = (long)S2 * Dm;
    g.M = NP; g.N = Dm; g.K = Dm; g.flags = GF_BIAS;
    launch_gemm(g, Bn, stream);
  }
  hipLaunchKernelGGL(k_cls, dim3(((long)Bn * NC * Dm + 255) / 256), dim3(256),
                     0, stream, xd, cls_emb);

  // decoder
  for (int l = 0; l < 6; ++l) {
    BlkW w = mk_blk(d_in, 21, l);
    run_block(xd, S2, w, xln, qbuf, kbuf, vT, attf, attb, hid,
              wqT, wkT, wvT, f1T, f2T, stream);
  }
  // final LN
  hipLaunchKernelGGL(k_ln, dim3((int)BS2), dim3(256), 0, stream, xd, dnw, dnb, xln, Dm);
  // pt = x[:, :196] @ proj_patch ; cf = x[:, 196:] @ proj_classes
  {
    GemmP g = gdef();
    g.A = xln; g.lda = Dm; g.sAb = (long)S2 * Dm;
    g.W = ppT; g.ldw = Dm;
    g.C = ptf; g.ldc = Dm; g.sCb = (long)NP * Dm;
    g.M = NP; g.N = Dm; g.K = Dm;
    launch_gemm(g, Bn, stream);
    g.A = xln + (long)NP * Dm; g.W = pcT;
    g.C = cff; g.sCb = (long)NC * Dm; g.M = NC;
    launch_gemm(g, Bn, stream);
  }
  hipLaunchKernelGGL(k_l2n, dim3(Bn * NP), dim3(256), 0, stream, ptf, ptn, Dm);
  hipLaunchKernelGGL(k_l2n, dim3(Bn * NC), dim3(256), 0, stream, cff, cfn, Dm);
  // masks = ptn @ cfn^T
  {
    GemmP g = gdef();
    g.A = ptn; g.lda = Dm; g.sAb = (long)NP * Dm;
    g.W = cfn; g.ldw = Dm; g.sWb = (long)NC * Dm;
    g.C = mskf; g.ldc = NC; g.sCb = (long)NP * NC;
    g.M = NP; g.N = NC; g.K = Dm;
    launch_gemm(g, Bn, stream);
  }
  hipLaunchKernelGGL(k_masknorm, dim3(((long)Bn * NP + 255) / 256), dim3(256),
                     0, stream, mskf, mnw, mnb, mskT);
  hipLaunchKernelGGL(k_resize, dim3(((long)Bn * NC * 224 * 224 + 255) / 256),
                     dim3(256), 0, stream, mskT, out);
}
